// CosineClassifier_77773267796467
// MI455X (gfx1250) — matrix operand order repaired and checked by reading, not compiled, not run on hardware
//
#include <hip/hip_runtime.h>

// CDNA5 / gfx1250: wave32, WMMA 16x16x32 f16 -> f32
typedef __attribute__((ext_vector_type(16))) _Float16 v16h;
typedef __attribute__((ext_vector_type(8)))  _Float16 h8;
typedef __attribute__((ext_vector_type(4)))  _Float16 h4;
typedef __attribute__((ext_vector_type(8)))  float    v8f;
typedef __attribute__((ext_vector_type(4)))  float    f4;

#define TEMP_F 0.05f
#define NEPS   1e-12f

// out[b,c] = dot(concept[b,c], w[b]) / max(||concept[b,c]||, eps)
// where w[b] = img[b] / (max(||img[b]||, eps) * TEMP)
//
// Per block: one batch b, 128 concept rows (8 waves x 16 rows).
// Per wave:  16 rows; K streamed in 32-wide chunks. Raw f32 is split into
// f16 hi/lo ONCE in registers at staging time; LDS holds ready-to-use f16
// tiles so fragments are plain ds_load_b128. Global loads for chunk k+1 are
// prefetched (NT, b128, coalesced) while chunk k is converted + WMMA'd.
__global__ __launch_bounds__(256)
void cosine_cls_wmma(const float* __restrict__ img,
                     const float* __restrict__ cpt,
                     float* __restrict__ out)
{
    constexpr int EMB  = 768;
    constexpr int NCLS = 4096;
    constexpr int LDSH = 48;   // padded LDS row stride in halves (32B-aligned rows)

    __shared__ __align__(32) _Float16 hi_tile[8][16][LDSH];
    __shared__ __align__(32) _Float16 lo_tile[8][16][LDSH];
    __shared__ __align__(32) _Float16 w_hi[EMB];
    __shared__ __align__(32) _Float16 w_lo[EMB];

    const int tid     = threadIdx.x;
    const int wv      = tid >> 5;
    const int lane    = tid & 31;
    const int b       = blockIdx.y;
    const int rowbase = blockIdx.x * 128 + wv * 16;

    // ---- wave 0: normalized, temperature-scaled image vector (hi/lo f16)
    if (wv == 0) {
        const float* irow = img + (size_t)b * EMB;
        float ss = 0.f;
#pragma unroll
        for (int j = 0; j < EMB / 32; ++j) {
            float x = irow[lane + 32 * j];
            ss = fmaf(x, x, ss);
        }
#pragma unroll
        for (int off = 16; off > 0; off >>= 1)
            ss += __shfl_xor(ss, off, 32);
        float scale = 1.0f / (fmaxf(sqrtf(ss), NEPS) * TEMP_F);
#pragma unroll
        for (int j = 0; j < EMB / 32; ++j) {
            int d = lane + 32 * j;
            float x = irow[d] * scale;
            _Float16 h = (_Float16)x;
            w_hi[d] = h;
            w_lo[d] = (_Float16)(x - (float)h);
        }
    }
    __syncthreads();

    const float* base = cpt + ((size_t)b * NCLS + rowbase) * EMB;

    v8f accD = {};   // dot(row, w), replicated across the 16 columns
    v8f accS = {};   // Gram accumulator: diag = sum of squares per row

    const int ha = (lane >> 4) * 16;   // A-fragment K base (contiguous K order, matches B)
    const int gb = (lane >> 4) * 16;   // B-fragment K base (contiguous K order, same as A)
    const int r  = lane & 15;
    const int rr = (lane >> 3) & 3;    // staging row within a 4-row pass
    const int cc = (lane & 7) * 4;     // staging column (floats)

    // ---- prologue: load chunk 0 into registers (coalesced 128B/row, NT)
    f4 cur[4];
#pragma unroll
    for (int p = 0; p < 4; ++p)
        cur[p] = __builtin_nontemporal_load(
            (const f4*)(base + (size_t)(p * 4 + rr) * EMB + cc));

    for (int kb = 0; kb < EMB; kb += 32) {
        // ---- prefetch chunk kb+32 while we convert/consume chunk kb
        f4 nxt[4];
        if (kb + 32 < EMB) {
#pragma unroll
            for (int p = 0; p < 4; ++p)
                nxt[p] = __builtin_nontemporal_load(
                    (const f4*)(base + (size_t)(p * 4 + rr) * EMB + (kb + 32) + cc));
        }

        // ---- split f32 -> f16 hi/lo once, stage f16 tiles to LDS (b64 stores)
#pragma unroll
        for (int p = 0; p < 4; ++p) {
            h4 hi, lo;
#pragma unroll
            for (int i = 0; i < 4; ++i) {
                float x = cur[p][i];
                _Float16 hx = (_Float16)x;
                hi[i] = hx;
                lo[i] = (_Float16)(x - (float)hx);
            }
            int row = p * 4 + rr;
            *(h4*)&hi_tile[wv][row][cc] = hi;
            *(h4*)&lo_tile[wv][row][cc] = lo;
        }
        // same-wave DS ops are in-order: no barrier needed (private tile).

        // ---- A fragments: row r, K in [ha,ha+16)
        h8 a0 = *(const h8*)&hi_tile[wv][r][ha];
        h8 a1 = *(const h8*)&hi_tile[wv][r][ha + 8];
        h8 l0 = *(const h8*)&lo_tile[wv][r][ha];
        h8 l1 = *(const h8*)&lo_tile[wv][r][ha + 8];
        v16h ahi, alo;
#pragma unroll
        for (int i = 0; i < 8; ++i) {
            ahi[i] = a0[i]; ahi[8 + i] = a1[i];
            alo[i] = l0[i]; alo[8 + i] = l1[i];
        }
        // ---- B fragments for Gram (B = A^T): column n=r, K in [gb, gb+16)
        v16h bqh = *(const v16h*)&hi_tile[wv][r][gb];
        v16h bql = *(const v16h*)&lo_tile[wv][r][gb];
        // ---- w fragments (replicated across all 16 columns)
        v16h bwh = *(const v16h*)&w_hi[kb + gb];
        v16h bwl = *(const v16h*)&w_lo[kb + gb];

        // dot = Ahi*Whi + Alo*Whi + Ahi*Wlo
        accD = __builtin_amdgcn_wmma_f32_16x16x32_f16(false, ahi, false, bwh,
                                                      (short)0, accD, false, false);
        accD = __builtin_amdgcn_wmma_f32_16x16x32_f16(false, alo, false, bwh,
                                                      (short)0, accD, false, false);
        accD = __builtin_amdgcn_wmma_f32_16x16x32_f16(false, ahi, false, bwl,
                                                      (short)0, accD, false, false);
        // sumsq(diag) = Ahi*Bhi + Alo*Bhi + Ahi*Blo
        accS = __builtin_amdgcn_wmma_f32_16x16x32_f16(false, ahi, false, bqh,
                                                      (short)0, accS, false, false);
        accS = __builtin_amdgcn_wmma_f32_16x16x32_f16(false, alo, false, bqh,
                                                      (short)0, accS, false, false);
        accS = __builtin_amdgcn_wmma_f32_16x16x32_f16(false, ahi, false, bql,
                                                      (short)0, accS, false, false);

#pragma unroll
        for (int p = 0; p < 4; ++p) cur[p] = nxt[p];
    }

    // ---- epilogue: row m's dot (col m) and diagonal sumsq both sit at lane
    // (m<8 ? m : m+16), accumulator vgpr m&7.
    if (lane < 8 || lane >= 24) {
        int m = (lane < 8) ? lane : (lane - 16);   // row 0..15
        int s = m & 7;
        float dot = accD[s];
        float ss  = accS[s];
        out[(size_t)b * NCLS + rowbase + m] = dot / fmaxf(sqrtf(ss), NEPS);
    }
}

extern "C" void kernel_launch(void* const* d_in, const int* in_sizes, int n_in,
                              void* d_out, int out_size, void* d_ws, size_t ws_size,
                              hipStream_t stream) {
    (void)in_sizes; (void)n_in; (void)d_ws; (void)ws_size; (void)out_size;
    const float* img = (const float*)d_in[0];   // (128, 768) f32
    const float* cpt = (const float*)d_in[1];   // (128, 4096, 768) f32
    float* out = (float*)d_out;                 // (128, 4096) f32

    dim3 grid(4096 / 128, 128);   // x: class tiles of 128, y: batch
    dim3 block(256);              // 8 waves (wave32), 16 rows per wave
    cosine_cls_wmma<<<grid, block, 0, stream>>>(img, cpt, out);
}
